// DeepFM_31112743092597
// MI455X (gfx1250) — matrix operand order repaired and checked by reading, not compiled, not run on hardware
//
#include <hip/hip_runtime.h>
#include <hip/hip_bf16.h>
#include <stdint.h>

#define B_      8192
#define NDENSE_ 13
#define NSPARSE_ 26
#define NFIELD_ 39
#define V_      40000
#define E_      64
#define H1_     512
#define H2_     256
#define K1_     (NFIELD_ * E_)   // 2496

typedef __attribute__((ext_vector_type(16))) __bf16    v16bf;
typedef __attribute__((ext_vector_type(8)))  float     v8f;
typedef __attribute__((ext_vector_type(4)))  unsigned  u32x4;
typedef __attribute__((ext_vector_type(4)))  unsigned  uint32x4_t;
typedef __attribute__((ext_vector_type(8)))  int       int32x8_t;
typedef __attribute__((ext_vector_type(4)))  int       int32x4_t;

// explicit global-address-space pointers so hot loops emit global_*, not flat_*
typedef const unsigned short __attribute__((address_space(1)))* gc_u16;
typedef unsigned short       __attribute__((address_space(1)))* g_u16;
typedef const float          __attribute__((address_space(1)))* gc_f32;
typedef float                __attribute__((address_space(1)))* g_f32;

union FragAB { u32x4 q[2]; v16bf v; };

__device__ __forceinline__ unsigned short f2bf(float f) {
  unsigned u = __float_as_uint(f);
  unsigned r = u + 0x7FFFu + ((u >> 16) & 1u);
  return (unsigned short)(r >> 16);
}
__device__ __forceinline__ float bf2f(unsigned short h) {
  return __uint_as_float(((unsigned)h) << 16);
}

#if __has_builtin(__builtin_amdgcn_tensor_load_to_lds)
#define USE_TDM 1
#else
#define USE_TDM 0
#endif

#if USE_TDM
// 2D TDM: tile 32 (K, contiguous) x 128 (rows), bf16, LDS pad 16B per 64B row
// (row stride in LDS = 80B -> conflict-free ds_load_b128 fragment reads).
__device__ __forceinline__ void tdm_load_tile_2d(unsigned lds_byte_addr,
                                                 const unsigned short* gsrc,
                                                 int rowlen, int nrows_total) {
  unsigned long long ga = (unsigned long long)(uintptr_t)gsrc;
  uint32x4_t g0;
  g0[0] = 1u;                                            // count=1, user descriptor
  g0[1] = lds_byte_addr;                                 // lds_addr
  g0[2] = (unsigned)ga;                                  // global_addr[31:0]
  g0[3] = (unsigned)((ga >> 32) & 0x1FFFFFFu) | (2u << 30);  // addr[56:32] | type=2
  int32x8_t g1;
  g1[0] = (int)((1u << 16) |                             // data_size = 2B
                (1u << 20) |                             // pad_enable
                (3u << 22) |                             // pad_interval: 16 dwords (64B)
                (3u << 25));                             // pad_amount: 4 dwords (16B)
  g1[1] = (int)(((unsigned)rowlen & 0xFFFFu) << 16);     // tensor_dim0[15:0]
  g1[2] = (int)((((unsigned)rowlen >> 16) & 0xFFFFu) |
                (((unsigned)nrows_total & 0xFFFFu) << 16));   // dim0 hi | dim1 lo
  g1[3] = (int)((((unsigned)nrows_total >> 16) & 0xFFFFu) |
                (32u << 16));                            // dim1 hi | tile_dim0=32
  g1[4] = 128;                                           // tile_dim1=128, tile_dim2=0
  g1[5] = rowlen;                                        // tensor_dim0_stride[31:0]
  g1[6] = 0;
  g1[7] = 0;
  int32x4_t gz4 = {0, 0, 0, 0};
  int32x8_t gz8 = {0, 0, 0, 0, 0, 0, 0, 0};
  // 6-arg form (clang-23 / therock-10.0 headers): g0, g1, g2, g3, g4, cpol
  __builtin_amdgcn_tensor_load_to_lds(g0, g1, gz4, gz4, gz8, 0);
}
#else
// Fallback: threads 0..127 each copy one 32-elem row of a 128x32 tile (pad stride 40 elems)
__device__ __forceinline__ void coop_load_tile(unsigned short* dst,
                                               const unsigned short* src, int K) {
  int t = threadIdx.x;
  if (t < 128) {
    const u32x4* s = (const u32x4*)(src + (size_t)t * K);
    u32x4* d = (u32x4*)(dst + t * 40);
    d[0] = s[0]; d[1] = s[1]; d[2] = s[2]; d[3] = s[3];
  }
}
#endif

// ---------------------------------------------------------------- utilities
__global__ void zero_kernel(float* p, int n) {
  int i = blockIdx.x * 256 + threadIdx.x;
  if (i < n) p[i] = 0.f;
}

// W1 (K1 x H1 fp32) -> W1bT (H1 x K1 bf16, transposed)
__global__ __launch_bounds__(256) void convert_w1(const float* __restrict__ W1,
                                                  unsigned short* __restrict__ W1bT) {
  int idx = blockIdx.x * 256 + threadIdx.x;
  if (idx >= H1_ * K1_) return;
  int n = idx / K1_;
  int k = idx - n * K1_;
  gc_f32 W1g = (gc_f32)(uintptr_t)W1;
  g_u16  Wtg = (g_u16)(uintptr_t)W1bT;
  Wtg[idx] = f2bf(W1g[(size_t)k * H1_ + n]);
}

// ---------------------------------------------------- gather + FM (1st/2nd)
__global__ __launch_bounds__(256) void embed_fm_kernel(
    const int* __restrict__ Xi, const float* __restrict__ Xv,
    const float* __restrict__ w1d, const float* __restrict__ b1d,
    const float* __restrict__ emb1,
    const float* __restrict__ w2d, const float* __restrict__ b2d,
    const float* __restrict__ emb2,
    unsigned short* __restrict__ deep, float* __restrict__ fmpart) {
  __shared__ float red[256];
  int tid = threadIdx.x;
  int sub = tid >> 6;          // row within block (4 rows / block)
  int e   = tid & 63;          // embedding lane
  int row = blockIdx.x * 4 + sub;
  const int*   xi = Xi + row * NFIELD_;
  const float* xv = Xv + row * NFIELD_;
  gc_f32 emb1g = (gc_f32)(uintptr_t)emb1;
  gc_f32 emb2g = (gc_f32)(uintptr_t)emb2;
  g_u16  drow  = (g_u16)(uintptr_t)(deep + (size_t)row * K1_);
  float s = 0.f, ssq = 0.f, fmacc = 0.f;
  for (int f = 0; f < NFIELD_; ++f) {
    float xvf = xv[f];
    float so, fo;
    if (f < NDENSE_) {
      float d = (float)xi[f];
      fo = (d * w1d[f * E_ + e] + b1d[f * E_ + e]) * xvf;
      so =  d * w2d[f * E_ + e] + b2d[f * E_ + e];          // no Xv on so_dense
    } else {
      int t = f - NDENSE_;
      int idx = xi[f];
      size_t base = ((size_t)t * V_ + (size_t)idx) * E_ + e;
      fo = emb1g[base] * xvf;
      so = emb2g[base] * xvf;
    }
    fmacc += fo;
    s     += so;
    ssq   += so * so;
    drow[f * E_ + e] = f2bf(so);
  }
  red[tid] = fmacc + 0.5f * (s * s - ssq);
  __syncthreads();
  for (int off = 32; off >= 1; off >>= 1) {
    if (e < off) red[tid] += red[tid + off];
    __syncthreads();
  }
  if (e == 0) fmpart[row] = red[sub << 6];
}

// -------------------------------------------------------- bf16 WMMA GEMM
// A: M x K bf16 row-major.  Bt: N x K bf16 (B transposed).
// TDM stages 128x32 A and B tiles into LDS (double buffered); 8 waves
// (4M x 2N) each compute a 32x64 WMMA tile from LDS fragments.
__global__ __launch_bounds__(256) void gemm_bf16_wmma(
    const unsigned short* __restrict__ A,
    const unsigned short* __restrict__ Bt,
    const float* __restrict__ bias,
    unsigned short* __restrict__ outb,
    float* __restrict__ outf,
    int M, int N, int K) {
  // layout (elements): buf b: A tile at b*10240, B tile at b*10240+5120
  __shared__ __align__(16) unsigned short smem[20480];
  int lane = threadIdx.x & 31;
  int wave = threadIdx.x >> 5;
  int wm = wave & 3;
  int wn = wave >> 2;
  int half = lane >> 4;
  int l16  = lane & 15;
  int blockM = blockIdx.x * 128;
  int blockN = blockIdx.y * 128;
  unsigned lds_base = (unsigned)(uintptr_t)(void*)smem;  // low 32 bits = LDS offset

  const unsigned short* Ablk = A  + (size_t)blockM * K;
  const unsigned short* Bblk = Bt + (size_t)blockN * K;

  v8f c[2][4] = {};

  // per-lane LDS fragment element offsets (row stride 40 elems = 80B)
  int aoffs[2], boffs[4];
#pragma unroll
  for (int mt = 0; mt < 2; ++mt)
    aoffs[mt] = (wm * 32 + mt * 16 + l16) * 40 + half * 8;
#pragma unroll
  for (int nt = 0; nt < 4; ++nt)
    boffs[nt] = (wn * 64 + nt * 16 + l16) * 40 + half * 8;

  const int kIters = K / 32;

  // prologue: fill buffer 0
#if USE_TDM
  if (wave == 0) {
    tdm_load_tile_2d(lds_base,          Ablk, K, M);
    tdm_load_tile_2d(lds_base + 10240u, Bblk, K, N);
  }
#else
  coop_load_tile(smem,        Ablk, K);
  coop_load_tile(smem + 5120, Bblk, K);
#endif

  for (int it = 0; it < kIters; ++it) {
    int cur = it & 1;
#if USE_TDM
    if (wave == 0) __builtin_amdgcn_s_wait_tensorcnt(0);
#endif
    __syncthreads();   // buf[cur] ready for all waves; prior reads of buf[cur^1] done
    if (it + 1 < kIters) {
      int kb = (it + 1) * 32;
#if USE_TDM
      if (wave == 0) {
        unsigned lb = lds_base + (unsigned)(cur ^ 1) * 20480u;
        tdm_load_tile_2d(lb,          Ablk + kb, K, M);
        tdm_load_tile_2d(lb + 10240u, Bblk + kb, K, N);
      }
#else
      coop_load_tile(smem + (cur ^ 1) * 10240,        Ablk + kb, K);
      coop_load_tile(smem + (cur ^ 1) * 10240 + 5120, Bblk + kb, K);
#endif
    }
    const unsigned short* sA = smem + cur * 10240;
    const unsigned short* sB = smem + cur * 10240 + 5120;
    FragAB a[2], b[4];
#pragma unroll
    for (int mt = 0; mt < 2; ++mt) {
      a[mt].q[0] = *(const u32x4*)(sA + aoffs[mt]);
      a[mt].q[1] = *(const u32x4*)(sA + aoffs[mt] + 16);
    }
#pragma unroll
    for (int nt = 0; nt < 4; ++nt) {
      b[nt].q[0] = *(const u32x4*)(sB + boffs[nt]);
      b[nt].q[1] = *(const u32x4*)(sB + boffs[nt] + 16);
    }
#pragma unroll
    for (int mt = 0; mt < 2; ++mt)
#pragma unroll
      for (int nt = 0; nt < 4; ++nt)
        c[mt][nt] = __builtin_amdgcn_wmma_f32_16x16x32_bf16(
            false, a[mt].v, false, b[nt].v, (short)0, c[mt][nt], false, false);
  }

  g_u16 outbg = (g_u16)(uintptr_t)outb;
  g_f32 outfg = (g_f32)(uintptr_t)outf;
#pragma unroll
  for (int mt = 0; mt < 2; ++mt) {
#pragma unroll
    for (int nt = 0; nt < 4; ++nt) {
      int n = blockN + wn * 64 + nt * 16 + l16;
      float bn = bias[n];
#pragma unroll
      for (int r = 0; r < 8; ++r) {
        int m = blockM + wm * 32 + mt * 16 + r + half * 8;
        float val = c[mt][nt][r] + bn;
        if (outb) outbg[(size_t)m * N + n] = f2bf(val);
        else      outfg[(size_t)m * N + n] = val;
      }
    }
  }
}

// ------------------------------------------------------- column statistics
__global__ __launch_bounds__(256) void colsum_bf16(const unsigned short* __restrict__ X,
                                                   int ncol, float* __restrict__ sum,
                                                   float* __restrict__ sq) {
  int tid = threadIdx.x;
  int row0 = blockIdx.x * 32;
  gc_u16 Xg = (gc_u16)(uintptr_t)X;
  for (int c = tid; c < ncol; c += 256) {
    float s = 0.f, q = 0.f;
    for (int r = 0; r < 32; ++r) {
      float v = bf2f(Xg[(size_t)(row0 + r) * ncol + c]);
      s += v; q += v * v;
    }
    atomicAdd(&sum[c], s);
    atomicAdd(&sq[c], q);
  }
}

__global__ __launch_bounds__(256) void colsum_f32(const float* __restrict__ X,
                                                  int ncol, float* __restrict__ sum,
                                                  float* __restrict__ sq) {
  int tid = threadIdx.x;
  int row0 = blockIdx.x * 32;
  gc_f32 Xg = (gc_f32)(uintptr_t)X;
  for (int c = tid; c < ncol; c += 256) {
    float s = 0.f, q = 0.f;
    for (int r = 0; r < 32; ++r) {
      float v = Xg[(size_t)(row0 + r) * ncol + c];
      s += v; q += v * v;
    }
    atomicAdd(&sum[c], s);
    atomicAdd(&sq[c], q);
  }
}

// BN1 fold: alpha = g1*rsqrt(var+eps), beta = be1 - alpha*mean
__global__ void finalize1(const float* sum, const float* sq, const float* g1,
                          const float* be1, float* alpha, float* beta) {
  int j = blockIdx.x * 256 + threadIdx.x;
  if (j < H1_) {
    float m = sum[j] * (1.f / B_);
    float var = sq[j] * (1.f / B_) - m * m;
    float a = g1[j] * rsqrtf(var + 1e-5f);
    alpha[j] = a;
    beta[j]  = be1[j] - a * m;
  }
}

// W2T'[n][j] = alpha_j * W2[j][n] (bf16, transposed);  bias2[n] = c2[n] + beta . W2[:,n]
__global__ __launch_bounds__(64) void prep_w2(const float* __restrict__ W2,
                                              const float* __restrict__ c2,
                                              const float* __restrict__ alpha,
                                              const float* __restrict__ beta,
                                              unsigned short* __restrict__ W2T,
                                              float* __restrict__ bias2) {
  __shared__ float red[64];
  int n = blockIdx.x;
  int tid = threadIdx.x;
  float bacc = 0.f;
  for (int j = tid; j < H1_; j += 64) {
    float w = W2[(size_t)j * H2_ + n];
    bacc += beta[j] * w;
    W2T[(size_t)n * H1_ + j] = f2bf(alpha[j] * w);
  }
  red[tid] = bacc; __syncthreads();
  for (int off = 32; off >= 1; off >>= 1) {
    if (tid < off) red[tid] += red[tid + off];
    __syncthreads();
  }
  if (tid == 0) bias2[n] = c2[n] + red[0];
}

// BN2 fold into row-dot vector a_k and scalar hconst
__global__ __launch_bounds__(256) void finalize2(const float* sum, const float* sq,
                                                 const float* g2, const float* be2,
                                                 float* avec, float* hconst) {
  __shared__ float red[256];
  int k = threadIdx.x;
  float m = sum[k] * (1.f / B_);
  float var = sq[k] * (1.f / B_) - m * m;
  float a = g2[k] * rsqrtf(var + 1e-5f);
  avec[k] = a;
  red[k] = be2[k] - a * m;
  __syncthreads();
  for (int off = 128; off >= 1; off >>= 1) {
    if (k < off) red[k] += red[k + off];
    __syncthreads();
  }
  if (k == 0) hconst[0] = red[0];
}

// out[row] = fmpart + bias + sum_k a_k * x2[row,k] + hconst
__global__ __launch_bounds__(256) void final_kernel(const float* __restrict__ x2,
                                                    const float* __restrict__ avec,
                                                    const float* __restrict__ hconst,
                                                    const float* __restrict__ fmpart,
                                                    const float* __restrict__ bias,
                                                    float* __restrict__ out) {
  int lane = threadIdx.x & 31;
  int wave = threadIdx.x >> 5;
  int row = blockIdx.x * 8 + wave;
  gc_f32 xr = (gc_f32)(uintptr_t)(x2 + (size_t)row * H2_);
  float acc = 0.f;
#pragma unroll
  for (int i = 0; i < 8; ++i) {
    int k = lane + i * 32;
    acc += avec[k] * xr[k];
  }
#pragma unroll
  for (int m = 16; m >= 1; m >>= 1) acc += __shfl_xor(acc, m, 32);
  if (lane == 0) out[row] = fmpart[row] + bias[row] + acc + hconst[0];
}

// --------------------------------------------------------------- launcher
extern "C" void kernel_launch(void* const* d_in, const int* in_sizes, int n_in,
                              void* d_out, int out_size, void* d_ws, size_t ws_size,
                              hipStream_t stream) {
  const int*   Xi   = (const int*)d_in[0];
  const float* Xv   = (const float*)d_in[1];
  const float* w1d  = (const float*)d_in[2];
  const float* b1d  = (const float*)d_in[3];
  const float* emb1 = (const float*)d_in[4];
  const float* w2d  = (const float*)d_in[5];
  const float* b2d  = (const float*)d_in[6];
  const float* emb2 = (const float*)d_in[7];
  const float* W1   = (const float*)d_in[8];
  const float* c1   = (const float*)d_in[9];
  const float* g1   = (const float*)d_in[10];
  const float* be1  = (const float*)d_in[11];
  const float* W2   = (const float*)d_in[12];
  const float* c2   = (const float*)d_in[13];
  const float* g2   = (const float*)d_in[14];
  const float* be2  = (const float*)d_in[15];
  const float* bias = (const float*)d_in[16];
  float* out = (float*)d_out;

  char* ws = (char*)d_ws;
  size_t off = 0;
  auto alloc = [&](size_t bytes) -> char* {
    char* p = ws + off;
    off = (off + bytes + 255) & ~(size_t)255;
    return p;
  };
  unsigned short* deep  = (unsigned short*)alloc((size_t)B_ * K1_ * 2);
  unsigned short* W1bT  = (unsigned short*)alloc((size_t)H1_ * K1_ * 2);
  unsigned short* x1bf  = (unsigned short*)alloc((size_t)B_ * H1_ * 2);
  unsigned short* W2T   = (unsigned short*)alloc((size_t)H2_ * H1_ * 2);
  float* x2     = (float*)alloc((size_t)B_ * H2_ * 4);
  float* fmpart = (float*)alloc((size_t)B_ * 4);
  float* stats  = (float*)alloc((512 + 512 + 256 + 256) * 4);  // s1sum,s1sq,s2sum,s2sq
  float* s1sum = stats;
  float* s1sq  = stats + 512;
  float* s2sum = stats + 1024;
  float* s2sq  = stats + 1280;
  float* alpha = (float*)alloc(H1_ * 4);
  float* beta  = (float*)alloc(H1_ * 4);
  float* bias2 = (float*)alloc(H2_ * 4);
  float* avec  = (float*)alloc(H2_ * 4);
  float* hconst = (float*)alloc(4);

  // 0) zero stat accumulators (deterministic each call)
  zero_kernel<<<(1536 + 255) / 256, 256, 0, stream>>>(stats, 1536);

  // 1) W1 -> bf16 transposed
  convert_w1<<<(H1_ * K1_ + 255) / 256, 256, 0, stream>>>(W1, W1bT);

  // 2) gather + FM + deep(bf16)
  embed_fm_kernel<<<B_ / 4, 256, 0, stream>>>(Xi, Xv, w1d, b1d, emb1,
                                              w2d, b2d, emb2, deep, fmpart);

  // 3) GEMM1: x1 = deep @ W1 + c1  (bf16 out)
  gemm_bf16_wmma<<<dim3(B_ / 128, H1_ / 128), 256, 0, stream>>>(
      deep, W1bT, c1, x1bf, (float*)nullptr, B_, H1_, K1_);

  // 4) BN1 stats + fold into W2
  colsum_bf16<<<B_ / 32, 256, 0, stream>>>(x1bf, H1_, s1sum, s1sq);
  finalize1<<<(H1_ + 255) / 256, 256, 0, stream>>>(s1sum, s1sq, g1, be1, alpha, beta);
  prep_w2<<<H2_, 64, 0, stream>>>(W2, c2, alpha, beta, W2T, bias2);

  // 5) GEMM2: x2 = BN1(x1) @ W2 + c2  via folded weights (f32 out)
  gemm_bf16_wmma<<<dim3(B_ / 128, H2_ / 128), 256, 0, stream>>>(
      x1bf, W2T, bias2, (unsigned short*)nullptr, x2, B_, H2_, H1_);

  // 6) BN2 stats folded into dot vector
  colsum_f32<<<B_ / 32, 256, 0, stream>>>(x2, H2_, s2sum, s2sq);
  finalize2<<<1, 256, 0, stream>>>(s2sum, s2sq, g2, be2, avec, hconst);

  // 7) final reduction per row
  final_kernel<<<B_ / 8, 256, 0, stream>>>(x2, avec, hconst, fmpart, bias, out);
}
